// LSTMReg_69114613729107
// MI455X (gfx1250) — compile-verified, matrix operand order checked by reading, not run on hardware
//
#include <hip/hip_runtime.h>

// ---------------------------------------------------------------------------
// LSTM tagger for MI455X (gfx1250, wave32).
//  Phase A: Gin[t, 4H] = sentence @ w_ih^T + (b_ih + b_hh)  -- bf16 WMMA GEMM
//  Phase B: 512 sequential fused step kernels: gates = Gin[t] + w_hh @ h
//           (bandwidth-bound matvec from L2; wave-per-hidden-unit, 4 gate
//            rows per wave so cell update needs no cross-block sync)
//  Phase C: y = w_out @ h + b_out; 10 AR steps reuse the step kernel with the
//           w_ih @ y matvec fused in.
// ---------------------------------------------------------------------------

typedef __bf16 bf16x16 __attribute__((ext_vector_type(16)));
typedef float f32x8   __attribute__((ext_vector_type(8)));

union Frag16 { bf16x16 v; uint4 q[2]; };

// ---------------- fp32 -> bf16 (round-to-nearest-even) ---------------------
__global__ void cvt_f32_to_bf16(const float* __restrict__ src,
                                unsigned short* __restrict__ dst, int n) {
    int i = blockIdx.x * blockDim.x + threadIdx.x;
    int stride = gridDim.x * blockDim.x;
    for (; i < n; i += stride) {
        unsigned int u = __float_as_uint(src[i]);
        u += 0x7FFFu + ((u >> 16) & 1u);
        dst[i] = (unsigned short)(u >> 16);
    }
}

__global__ void vec_add_f32(const float* __restrict__ a, const float* __restrict__ b,
                            float* __restrict__ o, int n) {
    int i = blockIdx.x * blockDim.x + threadIdx.x;
    if (i < n) o[i] = a[i] + b[i];
}

__global__ void zero_f32(float* __restrict__ p, int n) {
    int i = blockIdx.x * blockDim.x + threadIdx.x;
    if (i < n) p[i] = 0.f;
}

// ---------------- Phase A: bf16 WMMA GEMM  D[M,N] = A[M,K] * Bn[N,K]^T + bias
// A, Bn row-major bf16; D row-major fp32. One wave -> 32x32 output macro-tile.
// Fragment layout per CDNA5 ISA 7.12.2 (16-bit A 16x32 / B 32x16):
//   lane = r + 16*half ; per-lane K pattern = {half*8+0..7} u {16+half*8+0..7}
//   -> two 16B loads per fragment per lane from a row-major matrix row.
__global__ void __launch_bounds__(256)
gates_gemm_wmma(const unsigned short* __restrict__ A,
                const unsigned short* __restrict__ Bn,
                const float* __restrict__ bias,
                float* __restrict__ D,
                int M, int N, int K) {
    const int lane  = threadIdx.x & 31;
    const int wid   = blockIdx.x * (blockDim.x >> 5) + (threadIdx.x >> 5);
    const int nBlks = N >> 5;
    const int mb = wid / nBlks;
    const int nb = wid % nBlks;
    if (mb * 32 >= M) return;

    const int half = lane >> 4;   // 0..1
    const int r    = lane & 15;   // row-in-tile (A) / col-in-tile (B)
    const int m0 = mb * 32, n0 = nb * 32;

    const unsigned short* a0p = A  + (size_t)(m0 + r)      * K + half * 8;
    const unsigned short* a1p = A  + (size_t)(m0 + 16 + r) * K + half * 8;
    const unsigned short* b0p = Bn + (size_t)(n0 + r)      * K + half * 8;
    const unsigned short* b1p = Bn + (size_t)(n0 + 16 + r) * K + half * 8;

    f32x8 acc00 = {0.f,0.f,0.f,0.f,0.f,0.f,0.f,0.f};
    f32x8 acc01 = acc00, acc10 = acc00, acc11 = acc00;

    for (int k = 0; k < K; k += 32) {
        Frag16 a0, a1, b0, b1;
        a0.q[0] = *(const uint4*)(a0p + k);
        a0.q[1] = *(const uint4*)(a0p + k + 16);
        a1.q[0] = *(const uint4*)(a1p + k);
        a1.q[1] = *(const uint4*)(a1p + k + 16);
        b0.q[0] = *(const uint4*)(b0p + k);
        b0.q[1] = *(const uint4*)(b0p + k + 16);
        b1.q[0] = *(const uint4*)(b1p + k);
        b1.q[1] = *(const uint4*)(b1p + k + 16);
        acc00 = __builtin_amdgcn_wmma_f32_16x16x32_bf16(false, a0.v, false, b0.v,
                                                        (short)0, acc00, false, false);
        acc01 = __builtin_amdgcn_wmma_f32_16x16x32_bf16(false, a0.v, false, b1.v,
                                                        (short)0, acc01, false, false);
        acc10 = __builtin_amdgcn_wmma_f32_16x16x32_bf16(false, a1.v, false, b0.v,
                                                        (short)0, acc10, false, false);
        acc11 = __builtin_amdgcn_wmma_f32_16x16x32_bf16(false, a1.v, false, b1.v,
                                                        (short)0, acc11, false, false);
    }

    const float bias0 = bias ? bias[n0 + r]      : 0.f;
    const float bias1 = bias ? bias[n0 + 16 + r] : 0.f;
#pragma unroll
    for (int i = 0; i < 8; ++i) {
        const int mA = m0 + half * 8 + i;      // C/D layout: VGPR i -> M = i + 8*half
        float* rowA = D + (size_t)mA * N;
        float* rowB = rowA + (size_t)16 * N;   // second M tile
        rowA[n0 + r]      = acc00[i] + bias0;
        rowA[n0 + 16 + r] = acc01[i] + bias1;
        rowB[n0 + r]      = acc10[i] + bias0;
        rowB[n0 + 16 + r] = acc11[i] + bias1;
    }
}

// ---------------- bf16-row dot helpers -------------------------------------
__device__ __forceinline__ float dot8(uint4 wq, float4 a, float4 b) {
    float s;
    s  = __uint_as_float(wq.x << 16)          * a.x;
    s += __uint_as_float(wq.x & 0xFFFF0000u)  * a.y;
    s += __uint_as_float(wq.y << 16)          * a.z;
    s += __uint_as_float(wq.y & 0xFFFF0000u)  * a.w;
    s += __uint_as_float(wq.z << 16)          * b.x;
    s += __uint_as_float(wq.z & 0xFFFF0000u)  * b.y;
    s += __uint_as_float(wq.w << 16)          * b.z;
    s += __uint_as_float(wq.w & 0xFFFF0000u)  * b.w;
    return s;
}

__device__ __forceinline__ float sigmoidf_(float x) { return 1.f / (1.f + expf(-x)); }

// ---------------- Phase B/C: fused LSTM cell step --------------------------
// One wave per hidden unit j; computes all 4 gate rows (j, H+j, 2H+j, 3H+j)
// so the cell update is local to the wave. Gin != null: input projection was
// precomputed (scan). Gin == null: also fuse w_ih @ x (AR steps).
__global__ void __launch_bounds__(256)
lstm_step(const float* __restrict__ Gin,            // [4H] or null
          const float* __restrict__ biasSum,        // [4H] (used when Gin==null)
          const unsigned short* __restrict__ w_ih,  // bf16 [4H,E] (when Gin==null)
          const float* __restrict__ x,              // [E]        (when Gin==null)
          const unsigned short* __restrict__ w_hh,  // bf16 [4H,H]
          const float* __restrict__ h_in,           // [H]
          float* __restrict__ c,                    // [H] in/out (owned per-j)
          float* __restrict__ h_out,                // [H] (double-buffered)
          int H, int E) {
    const int lane = threadIdx.x & 31;
    const int j = blockIdx.x * (blockDim.x >> 5) + (threadIdx.x >> 5);
    if (j >= H) return;

    const unsigned short* w0 = w_hh + (size_t)(0 * H + j) * H;
    const unsigned short* w1 = w_hh + (size_t)(1 * H + j) * H;
    const unsigned short* w2 = w_hh + (size_t)(2 * H + j) * H;
    const unsigned short* w3 = w_hh + (size_t)(3 * H + j) * H;

    float s0 = 0.f, s1 = 0.f, s2 = 0.f, s3 = 0.f;
    for (int k = lane * 8; k < H; k += 256) {          // 8 bf16 per lane per iter
        const float4* hp = (const float4*)(h_in + k);
        float4 ha = hp[0], hb = hp[1];
        s0 += dot8(*(const uint4*)(w0 + k), ha, hb);
        s1 += dot8(*(const uint4*)(w1 + k), ha, hb);
        s2 += dot8(*(const uint4*)(w2 + k), ha, hb);
        s3 += dot8(*(const uint4*)(w3 + k), ha, hb);
    }
    if (Gin == nullptr) {                              // AR path: fuse w_ih @ x
        const unsigned short* u0 = w_ih + (size_t)(0 * H + j) * E;
        const unsigned short* u1 = w_ih + (size_t)(1 * H + j) * E;
        const unsigned short* u2 = w_ih + (size_t)(2 * H + j) * E;
        const unsigned short* u3 = w_ih + (size_t)(3 * H + j) * E;
        for (int k = lane * 8; k < E; k += 256) {
            const float4* xp = (const float4*)(x + k);
            float4 xa = xp[0], xb = xp[1];
            s0 += dot8(*(const uint4*)(u0 + k), xa, xb);
            s1 += dot8(*(const uint4*)(u1 + k), xa, xb);
            s2 += dot8(*(const uint4*)(u2 + k), xa, xb);
            s3 += dot8(*(const uint4*)(u3 + k), xa, xb);
        }
    }
#pragma unroll
    for (int off = 16; off; off >>= 1) {
        s0 += __shfl_xor(s0, off, 32);
        s1 += __shfl_xor(s1, off, 32);
        s2 += __shfl_xor(s2, off, 32);
        s3 += __shfl_xor(s3, off, 32);
    }
    if (lane == 0) {
        float gi = s0, gf = s1, gg = s2, go = s3;
        if (Gin) { gi += Gin[j]; gf += Gin[H + j]; gg += Gin[2 * H + j]; go += Gin[3 * H + j]; }
        else     { gi += biasSum[j]; gf += biasSum[H + j]; gg += biasSum[2 * H + j]; go += biasSum[3 * H + j]; }
        float iv = sigmoidf_(gi);
        float fv = sigmoidf_(gf);
        float gv = tanhf(gg);
        float ov = sigmoidf_(go);
        float cn = fv * c[j] + iv * gv;
        c[j] = cn;
        h_out[j] = ov * tanhf(cn);
    }
}

// ---------------- y = W @ v + b  (wave per output row) ---------------------
__global__ void __launch_bounds__(256)
matvec_out(const unsigned short* __restrict__ W,  // bf16 [N,H]
           const float* __restrict__ v,           // [H]
           const float* __restrict__ bias,        // [N]
           float* __restrict__ y, int N, int H) {
    const int lane = threadIdx.x & 31;
    const int n = blockIdx.x * (blockDim.x >> 5) + (threadIdx.x >> 5);
    if (n >= N) return;
    const unsigned short* wr = W + (size_t)n * H;
    float s = 0.f;
    for (int k = lane * 8; k < H; k += 256) {
        const float4* hp = (const float4*)(v + k);
        s += dot8(*(const uint4*)(wr + k), hp[0], hp[1]);
    }
#pragma unroll
    for (int off = 16; off; off >>= 1) s += __shfl_xor(s, off, 32);
    if (lane == 0) y[n] = s + bias[n];
}

// ---------------------------------------------------------------------------
extern "C" void kernel_launch(void* const* d_in, const int* in_sizes, int n_in,
                              void* d_out, int out_size, void* d_ws, size_t ws_size,
                              hipStream_t stream) {
    (void)in_sizes; (void)n_in; (void)out_size; (void)ws_size;
    const int S = 512, E = 2048, H = 2048, T = 2048;
    const int G4 = 4 * H;  // 8192

    const float* sentence = (const float*)d_in[0];
    const float* w_ih     = (const float*)d_in[1];
    const float* w_hh     = (const float*)d_in[2];
    const float* b_ih     = (const float*)d_in[3];
    const float* b_hh     = (const float*)d_in[4];
    const float* w_out    = (const float*)d_in[5];
    const float* b_out    = (const float*)d_in[6];
    float* out = (float*)d_out;

    // Workspace carve-out (~91 MB), 256B-aligned slabs.
    char* ws = (char*)d_ws;
    size_t off = 0;
    auto alloc = [&](size_t bytes) -> void* {
        void* p = (void*)(ws + off);
        off += (bytes + 255) & ~(size_t)255;
        return p;
    };
    unsigned short* sent_bf = (unsigned short*)alloc((size_t)S  * E * 2);
    unsigned short* wih_bf  = (unsigned short*)alloc((size_t)G4 * E * 2);
    unsigned short* whh_bf  = (unsigned short*)alloc((size_t)G4 * H * 2);
    unsigned short* wout_bf = (unsigned short*)alloc((size_t)T  * H * 2);
    float* Gin     = (float*)alloc((size_t)S * G4 * 4);
    float* biasSum = (float*)alloc((size_t)G4 * 4);
    float* state   = (float*)alloc((size_t)3 * H * 4);  // h0 | h1 | c

    auto cdiv = [](long long a, long long b) { return (int)((a + b - 1) / b); };
    const int THR = 256;

    // Convert operands to bf16 once per call.
    cvt_f32_to_bf16<<<cdiv((long long)S * E,  THR), THR, 0, stream>>>(sentence, sent_bf, S * E);
    cvt_f32_to_bf16<<<cdiv((long long)G4 * E, THR), THR, 0, stream>>>(w_ih,  wih_bf,  G4 * E);
    cvt_f32_to_bf16<<<cdiv((long long)G4 * H, THR), THR, 0, stream>>>(w_hh,  whh_bf,  G4 * H);
    cvt_f32_to_bf16<<<cdiv((long long)T * H,  THR), THR, 0, stream>>>(w_out, wout_bf, T * H);
    vec_add_f32<<<cdiv(G4, THR), THR, 0, stream>>>(b_ih, b_hh, biasSum, G4);
    zero_f32<<<cdiv(3 * H, THR), THR, 0, stream>>>(state, 3 * H);

    // Phase A: Gin = sentence @ w_ih^T + biasSum   (WMMA, 32x32 tile per wave)
    {
        const int waves = (S / 32) * (G4 / 32);     // 4096
        gates_gemm_wmma<<<waves / 8, THR, 0, stream>>>(sent_bf, wih_bf, biasSum,
                                                       Gin, S, G4, E);
    }

    // Phase B: 512 sequential cell steps (stream order is the sync).
    float* c = state + 2 * H;
    int p = 0;                                      // h lives in state + p*H
    for (int t = 0; t < S; ++t) {
        lstm_step<<<H / 8, THR, 0, stream>>>(Gin + (size_t)t * G4, nullptr, nullptr, nullptr,
                                             whh_bf, state + p * H, c, state + (1 - p) * H,
                                             H, E);
        p = 1 - p;
    }

    // Phase C: first output + 10 autoregressive steps.
    matvec_out<<<T / 8, THR, 0, stream>>>(wout_bf, state + p * H, b_out, out, T, H);
    for (int i = 0; i < 10; ++i) {
        lstm_step<<<H / 8, THR, 0, stream>>>(nullptr, biasSum, wih_bf, out + (size_t)i * T,
                                             whh_bf, state + p * H, c, state + (1 - p) * H,
                                             H, E);
        p = 1 - p;
        matvec_out<<<T / 8, THR, 0, stream>>>(wout_bf, state + p * H, b_out,
                                              out + (size_t)(i + 1) * T, T, H);
    }
}
